// TokenRoutedMLP_39067022524585
// MI455X (gfx1250) — matrix operand order repaired and checked by reading, not compiled, not run on hardware
//
#include <hip/hip_runtime.h>
#include <hip/hip_bf16.h>

#define DIM    1024
#define NEXP   16
#define INTER  256
#define NTOK   16384    // 4 * 4096
#define CHUNK  1024     // tokens per expert
#define MT     64       // token rows per workgroup

typedef __attribute__((ext_vector_type(16))) __bf16 v16bf;
typedef __attribute__((ext_vector_type(8)))  __bf16 v8bf;
typedef __attribute__((ext_vector_type(8)))  float  v8f;

#define LDH 264   // ldsH row stride (256 + 8 pad), 528B: multiple of 16B

// Build a 16-element bf16 fragment from two 8-element contiguous 16B chunks.
__device__ __forceinline__ v16bf frag16(const __bf16* lo, const __bf16* hi) {
  v8bf a = *(const v8bf*)lo;
  v8bf b = *(const v8bf*)hi;
  v16bf r;
#pragma unroll
  for (int i = 0; i < 8; ++i) { r[i] = a[i]; r[i + 8] = b[i]; }
  return r;
}

__device__ __forceinline__ void zero8(v8f& v) {
#pragma unroll
  for (int i = 0; i < 8; ++i) v[i] = 0.0f;
}

// ---------------- pre-pass 1: x f32 -> bf16, straight copy ----------------
__global__ __launch_bounds__(256)
void cvt_x(const float* __restrict__ in, __bf16* __restrict__ out) {
  int i = (blockIdx.x * 256 + threadIdx.x) * 8;
  float4 a = *(const float4*)(in + i);
  float4 b = *(const float4*)(in + i + 4);
  v8bf o;
  o[0] = (__bf16)a.x; o[1] = (__bf16)a.y; o[2] = (__bf16)a.z; o[3] = (__bf16)a.w;
  o[4] = (__bf16)b.x; o[5] = (__bf16)b.y; o[6] = (__bf16)b.z; o[7] = (__bf16)b.w;
  *(v8bf*)(out + i) = o;
}

// -------- pre-pass 2: per-expert transpose + convert: [R][C]f32 -> [C][R]bf16
__global__ __launch_bounds__(256)
void transpose_cvt(const float* __restrict__ in, __bf16* __restrict__ out,
                   int R, int C) {
  __shared__ __bf16 t[32][34];
  const int mat = blockIdx.z;
  const int rb = blockIdx.y * 32, cb = blockIdx.x * 32;
  const float* ip = in  + (size_t)mat * R * C;
  __bf16*      op = out + (size_t)mat * R * C;
  const int tx = threadIdx.x & 31, ty = threadIdx.x >> 5;
#pragma unroll
  for (int i = ty; i < 32; i += 8)
    t[i][tx] = (__bf16)ip[(size_t)(rb + i) * C + cb + tx];
  __syncthreads();
#pragma unroll
  for (int i = ty; i < 32; i += 8)
    op[(size_t)(cb + i) * R + rb + tx] = t[tx][i];
}

// ---------------- main fused MoE MLP kernel -------------------------------
__global__ __launch_bounds__(256)
void moe_mlp_fused(const __bf16* __restrict__ xbf,      // [N, DIM] bf16
                   const int*    __restrict__ sort_idx, // [N]
                   const __bf16* __restrict__ w1t,      // [E, INTER, DIM] bf16
                   const __bf16* __restrict__ w2t,      // [E, DIM, INTER] bf16
                   float*        __restrict__ out) {    // [N, DIM] f32
  __shared__ __bf16 ldsH[MT * LDH];  // h tile [64][256]
  __shared__ int    sIdx[MT];

  const int tid  = threadIdx.x;
  const int lane = tid & 31;
  const int wv   = tid >> 5;

  const int e    = blockIdx.x >> 4;       // expert
  const int mblk = blockIdx.x & 15;       // 64-row block within expert chunk
  const int base = e * CHUNK + mblk * MT; // sorted-row base

  if (tid < MT) sIdx[tid] = sort_idx[base + tid];
  __syncthreads();

  const int lhalf = lane >> 4;            // 0/1
  const int lmod  = lane & 15;

  // ---------------- GEMM1: h = relu(x @ W1)^2  (64 x 256) -----------------
  const int rg = wv >> 2;                 // 32-row group (0..1)
  const int cg = wv & 3;                  // 64-col group (0..3)
  v8f acc1[2][4];
#pragma unroll
  for (int i = 0; i < 2; ++i)
#pragma unroll
    for (int j = 0; j < 4; ++j) zero8(acc1[i][j]);

  // Per-lane A row pointers (gathered) and B column pointers, hoisted.
  const __bf16* arow[2];
#pragma unroll
  for (int mt = 0; mt < 2; ++mt)
    arow[mt] = xbf + (size_t)sIdx[rg * 32 + mt * 16 + lmod] * DIM + lhalf * 8;
  const __bf16* bcol[4];
#pragma unroll
  for (int nt = 0; nt < 4; ++nt)
    bcol[nt] = w1t + ((size_t)e * INTER + cg * 64 + nt * 16 + lmod) * DIM
                   + lhalf * 8;

#pragma unroll 2
  for (int kb = 0; kb < DIM; kb += 32) {
    v16bf afr[2], bfr[4];
#pragma unroll
    for (int mt = 0; mt < 2; ++mt)
      afr[mt] = frag16(arow[mt] + kb, arow[mt] + kb + 16);
#pragma unroll
    for (int nt = 0; nt < 4; ++nt)
      bfr[nt] = frag16(bcol[nt] + kb, bcol[nt] + kb + 16);
    if (kb + 64 < DIM) {
      __builtin_prefetch(arow[0] + kb + 64, 0, 1);
      __builtin_prefetch(bcol[0] + kb + 64, 0, 1);
    }
#pragma unroll
    for (int mt = 0; mt < 2; ++mt)
#pragma unroll
      for (int nt = 0; nt < 4; ++nt)
        acc1[mt][nt] = __builtin_amdgcn_wmma_f32_16x16x32_bf16(
            false, afr[mt], false, bfr[nt], (short)0, acc1[mt][nt], false, false);
  }

  // epilogue 1: h = relu(acc)^2 -> ldsH (bf16)
#pragma unroll
  for (int mt = 0; mt < 2; ++mt)
#pragma unroll
    for (int nt = 0; nt < 4; ++nt)
#pragma unroll
      for (int r = 0; r < 8; ++r) {
        int m = rg * 32 + mt * 16 + r + lhalf * 8;    // C/D lane layout
        int n = cg * 64 + nt * 16 + lmod;
        float v = acc1[mt][nt][r];
        v = v > 0.0f ? v * v : 0.0f;
        ldsH[m * LDH + n] = (__bf16)v;
      }
  __syncthreads();

  // ---------------- GEMM2: out = h @ W2  (64 x 1024) + scatter ------------
  const int rg2 = wv >> 2;   // 32-row group (0..1)
  const int cg2 = wv & 3;    // 32-col group inside 128-col chunk (0..3)

  // Scatter row indices for this lane, hoisted to registers.
  int orow[2][8];
#pragma unroll
  for (int mt = 0; mt < 2; ++mt)
#pragma unroll
    for (int r = 0; r < 8; ++r)
      orow[mt][r] = sIdx[rg2 * 32 + mt * 16 + lhalf * 8 + r];

  const __bf16* hrow[2];
#pragma unroll
  for (int mt = 0; mt < 2; ++mt)
    hrow[mt] = &ldsH[(rg2 * 32 + mt * 16 + lmod) * LDH + lhalf * 8];

  const __bf16* w2e = w2t + (size_t)e * DIM * INTER;

#pragma unroll 1
  for (int nc = 0; nc < DIM; nc += 128) {
    v8f acc2[2][2];
#pragma unroll
    for (int i = 0; i < 2; ++i)
#pragma unroll
      for (int j = 0; j < 2; ++j) zero8(acc2[i][j]);

    const __bf16* bcol2[2];
#pragma unroll
    for (int nt = 0; nt < 2; ++nt)
      bcol2[nt] = w2e + (size_t)(nc + cg2 * 32 + nt * 16 + lmod) * INTER
                      + lhalf * 8;

#pragma unroll 2
    for (int kb = 0; kb < INTER; kb += 32) {
      v16bf afr[2], bfr[2];
#pragma unroll
      for (int mt = 0; mt < 2; ++mt)
        afr[mt] = frag16(hrow[mt] + kb, hrow[mt] + kb + 16);
#pragma unroll
      for (int nt = 0; nt < 2; ++nt)
        bfr[nt] = frag16(bcol2[nt] + kb, bcol2[nt] + kb + 16);
#pragma unroll
      for (int mt = 0; mt < 2; ++mt)
#pragma unroll
        for (int nt = 0; nt < 2; ++nt)
          acc2[mt][nt] = __builtin_amdgcn_wmma_f32_16x16x32_bf16(
              false, afr[mt], false, bfr[nt], (short)0, acc2[mt][nt], false, false);
    }

    // scatter-store f32 results
#pragma unroll
    for (int mt = 0; mt < 2; ++mt)
#pragma unroll
      for (int nt = 0; nt < 2; ++nt) {
        int n = nc + cg2 * 32 + nt * 16 + lmod;
#pragma unroll
        for (int r = 0; r < 8; ++r)
          out[(size_t)orow[mt][r] * DIM + n] = acc2[mt][nt][r];
      }
  }
}

extern "C" void kernel_launch(void* const* d_in, const int* in_sizes, int n_in,
                              void* d_out, int out_size, void* d_ws, size_t ws_size,
                              hipStream_t stream) {
  const float* x    = (const float*)d_in[0];
  const int*   sidx = (const int*)d_in[1];
  const float* w1   = (const float*)d_in[2];
  const float* w2   = (const float*)d_in[3];
  float*       out  = (float*)d_out;

  // Workspace layout (bf16 side buffers): xbf 32MB | w1t 8MB | w2t 8MB
  char* ws = (char*)d_ws;
  __bf16* xbf = (__bf16*)ws;
  __bf16* w1t = (__bf16*)(ws + (size_t)32 * 1024 * 1024);
  __bf16* w2t = (__bf16*)(ws + (size_t)40 * 1024 * 1024);

  // pre-pass: convert x, transpose+convert weights
  hipLaunchKernelGGL(cvt_x, dim3(NTOK * DIM / (256 * 8)), dim3(256), 0, stream,
                     x, xbf);
  hipLaunchKernelGGL(transpose_cvt, dim3(INTER / 32, DIM / 32, NEXP), dim3(256),
                     0, stream, w1, w1t, DIM, INTER);   // -> [E][INTER][DIM]
  hipLaunchKernelGGL(transpose_cvt, dim3(DIM / 32, INTER / 32, NEXP), dim3(256),
                     0, stream, w2, w2t, INTER, DIM);   // -> [E][DIM][INTER]

  // main fused kernel: 16 experts * 16 row-blocks = 256 WGs, 8 waves each
  hipLaunchKernelGGL(moe_mlp_fused, dim3(NEXP * (CHUNK / MT)), dim3(256), 0,
                     stream, xbf, sidx, w1t, w2t, out);
}
